// LSTM_33852932227209
// MI455X (gfx1250) — matrix operand order repaired and checked by reading, not compiled, not run on hardware
//
#include <hip/hip_runtime.h>
#include <hip/hip_bf16.h>
#include <stddef.h>

// ---------------- problem constants ----------------
#define B_   128
#define T_   256
#define D_   256
#define H_   1024
#define KTOT (D_ + H_)          // 1280 (fused [x_t | h] contraction dim)
#define KHLF 640                // K split point per wave pair
#define NWG  128                // persistent workgroups
#define WGT  128                // threads per workgroup (4 waves = 2 pairs)

// ---------------- workspace layout (bytes) ----------------
#define OFF_CNT 0
#define OFF_HB  1024
#define OFF_HF  (OFF_HB + (size_t)B_ * H_ * 2)
#define OFF_XB  (OFF_HF + (size_t)B_ * H_ * 4)
#define OFF_WT  (OFF_XB + (size_t)T_ * B_ * D_ * 2)
// end = OFF_WT + 4096*1280*2  (~28.1 MB total)

typedef __attribute__((ext_vector_type(16))) __bf16 bf16x16;
typedef __attribute__((ext_vector_type(8)))  __bf16 bf16x8;
typedef __attribute__((ext_vector_type(8)))  float  floatx8;

__device__ __forceinline__ float sigmoid_(float x) {
  return 1.0f / (1.0f + __expf(-x));
}

// ---------------- prep: zero h + barrier counter ----------------
__global__ void k_zero(unsigned* __restrict__ cnt, __bf16* __restrict__ hb) {
  size_t i = (size_t)blockIdx.x * blockDim.x + threadIdx.x;
  if (i == 0) *cnt = 0u;
  if (i < (size_t)B_ * H_) hb[i] = (__bf16)0.0f;
}

// ---------------- prep: x [B,T,D] f32 -> xb [T,B,D] bf16 ----------------
__global__ void k_convx(const float* __restrict__ x, __bf16* __restrict__ xb) {
  size_t i = (size_t)blockIdx.x * blockDim.x + threadIdx.x;
  if (i >= (size_t)T_ * B_ * D_) return;
  int d = (int)(i % D_);
  int b = (int)((i / D_) % B_);
  int t = (int)(i / ((size_t)D_ * B_));
  xb[i] = (__bf16)x[((size_t)b * T_ + t) * D_ + d];
}

// ---------------- prep: Wt[n][k] = bf16( W[k][n] ), W = [Wx;Wh] ----------------
__global__ void k_buildWt(const float* __restrict__ Wfx, const float* __restrict__ Wix,
                          const float* __restrict__ Wgx, const float* __restrict__ Wox,
                          const float* __restrict__ Wfh, const float* __restrict__ Wih,
                          const float* __restrict__ Wgh, const float* __restrict__ Woh,
                          __bf16* __restrict__ Wt) {
  size_t i = (size_t)blockIdx.x * blockDim.x + threadIdx.x;
  if (i >= (size_t)4 * H_ * KTOT) return;
  int k  = (int)(i % KTOT);
  int nn = (int)(i / KTOT);
  int gi = nn >> 10;
  int jj = nn & (H_ - 1);
  const float* wx[4] = {Wfx, Wix, Wgx, Wox};
  const float* wh[4] = {Wfh, Wih, Wgh, Woh};
  float v = (k < D_) ? wx[gi][(size_t)k * H_ + jj]
                     : wh[gi][(size_t)(k - D_) * H_ + jj];
  Wt[i] = (__bf16)v;
}

// ---- K-half inner loop: 20 k-tiles, 2 M-tiles x 4 gates.
//      B fragments come from the register-resident bw[] array (no loads).
//      A loads software-pipelined one k-tile ahead. ----
template <int KHALF>
__device__ __forceinline__ void kloop(const __bf16* __restrict__ xA, const __bf16* __restrict__ xB,
                                      const __bf16* __restrict__ hA, const __bf16* __restrict__ hB,
                                      const bf16x16* bw,   // [4][20] flattened gi*20+kt
                                      int hoff, floatx8* acc) {
  const int KB = KHALF * KHLF;
  bf16x8 nl0, nh0, nl1, nh1;
  {
    const __bf16* pa = (KB < D_) ? (xA + KB) : (hA + (KB - D_));
    const __bf16* pb = (KB < D_) ? (xB + KB) : (hB + (KB - D_));
    nl0 = *(const bf16x8*)(pa + hoff);      nh0 = *(const bf16x8*)(pa + 8 + hoff);
    nl1 = *(const bf16x8*)(pb + hoff);      nh1 = *(const bf16x8*)(pb + 8 + hoff);
  }
#pragma unroll
  for (int kt = 0; kt < 20; ++kt) {
    bf16x8 cl0 = nl0, ch0 = nh0, cl1 = nl1, ch1 = nh1;
    if (kt < 19) {  // prefetch next k-tile's A fragments (1 iteration of distance)
      const int kg = KB + (kt + 1) * 32;
      const __bf16* pa = (kg < D_) ? (xA + kg) : (hA + (kg - D_));
      const __bf16* pb = (kg < D_) ? (xB + kg) : (hB + (kg - D_));
      nl0 = *(const bf16x8*)(pa + hoff);    nh0 = *(const bf16x8*)(pa + 8 + hoff);
      nl1 = *(const bf16x8*)(pb + hoff);    nh1 = *(const bf16x8*)(pb + 8 + hoff);
    }
    bf16x16 af0 = __builtin_shufflevector(cl0, ch0, 0,1,2,3,4,5,6,7,8,9,10,11,12,13,14,15);
    bf16x16 af1 = __builtin_shufflevector(cl1, ch1, 0,1,2,3,4,5,6,7,8,9,10,11,12,13,14,15);
    const bf16x16 b0 = bw[0 * 20 + kt];
    const bf16x16 b1 = bw[1 * 20 + kt];
    const bf16x16 b2 = bw[2 * 20 + kt];
    const bf16x16 b3 = bw[3 * 20 + kt];
    acc[0] = __builtin_amdgcn_wmma_f32_16x16x32_bf16(false, af0, false, b0, (short)0, acc[0], false, false);
    acc[1] = __builtin_amdgcn_wmma_f32_16x16x32_bf16(false, af1, false, b0, (short)0, acc[1], false, false);
    acc[2] = __builtin_amdgcn_wmma_f32_16x16x32_bf16(false, af0, false, b1, (short)0, acc[2], false, false);
    acc[3] = __builtin_amdgcn_wmma_f32_16x16x32_bf16(false, af1, false, b1, (short)0, acc[3], false, false);
    acc[4] = __builtin_amdgcn_wmma_f32_16x16x32_bf16(false, af0, false, b2, (short)0, acc[4], false, false);
    acc[5] = __builtin_amdgcn_wmma_f32_16x16x32_bf16(false, af1, false, b2, (short)0, acc[5], false, false);
    acc[6] = __builtin_amdgcn_wmma_f32_16x16x32_bf16(false, af0, false, b3, (short)0, acc[6], false, false);
    acc[7] = __builtin_amdgcn_wmma_f32_16x16x32_bf16(false, af1, false, b3, (short)0, acc[7], false, false);
  }
}

// ---------------- persistent LSTM recurrence ----------------
// 128 WGs x 4 waves.  WG: hidden slice j0 = 16*(g>>1), row half rh = g&1.
// Wave w: pair p = w>>1 handles rows m0 = 64*rh + 32*p (2 M-tiles);
//         khalf = w&1 handles K in [khalf*640, khalf*640+640).
// All B fragments per wave (4 gates x 20 ktiles x 8 VGPRs = 640 VGPRs) are
// explicitly preloaded into registers BEFORE the time loop -> weight-stationary.
// Wave pair reduces partial gate sums through LDS once per step.
__global__ __launch_bounds__(WGT, 1) void k_lstm(
    const __bf16* __restrict__ xb, const __bf16* __restrict__ Wt,
    const float* __restrict__ bfv, const float* __restrict__ biv,
    const float* __restrict__ bgv, const float* __restrict__ bov,
    __bf16* __restrict__ hb, float* __restrict__ hf, unsigned* __restrict__ cnt) {
  const int lane  = threadIdx.x & 31;
  const int wid   = threadIdx.x >> 5;     // 0..3
  const int p     = wid >> 1;             // pair 0..1
  const int khalf = wid & 1;              // K half
  const int g     = blockIdx.x;
  const int j0    = (g >> 1) << 4;        // 16 hidden cols
  const int rh    = g & 1;
  const int m0    = (rh << 6) + (p << 5); // 32-row block (2 M-tiles)
  const int n     = lane & 15;
  const int half  = lane >> 4;
  const int hoff  = half << 4;            // A K-subrange select
  const int half16 = half << 4;           // B K-subrange select
  const int mrowA = m0 + n;
  const int mrowB = mrowA + 16;

  const __bf16* w0 = Wt + (size_t)(0 * H_ + j0 + n) * KTOT;
  const __bf16* w1 = Wt + (size_t)(1 * H_ + j0 + n) * KTOT;
  const __bf16* w2 = Wt + (size_t)(2 * H_ + j0 + n) * KTOT;
  const __bf16* w3 = Wt + (size_t)(3 * H_ + j0 + n) * KTOT;

  // ---- explicit weight preload: registers hold this wave's entire B slice ----
  bf16x16 bw[80];
  {
    const int KB = khalf * KHLF;
#pragma unroll
    for (int kt = 0; kt < 20; ++kt) {
      const int kg = KB + kt * 32 + half16;
      bw[0 * 20 + kt] = *(const bf16x16*)(w0 + kg);
      bw[1 * 20 + kt] = *(const bf16x16*)(w1 + kg);
      bw[2 * 20 + kt] = *(const bf16x16*)(w2 + kg);
      bw[3 * 20 + kt] = *(const bf16x16*)(w3 + kg);
    }
  }

  const float bias0 = bfv[j0 + n];
  const float bias1 = biv[j0 + n];
  const float bias2 = bgv[j0 + n];
  const float bias3 = bov[j0 + n];

  __shared__ float xch[2][8][256];        // [pair][tile][lane*8] partial sums (16 KB)

  float cst[2][8];
#pragma unroll
  for (int mt = 0; mt < 2; ++mt)
#pragma unroll
    for (int r = 0; r < 8; ++r) cst[mt][r] = 0.0f;

  const __bf16* hA = hb + (size_t)mrowA * H_;
  const __bf16* hB = hb + (size_t)mrowB * H_;
  const __bf16* xA = xb + (size_t)mrowA * D_;   // advances by B_*D_ per step
  const __bf16* xB = xb + (size_t)mrowB * D_;

  for (int t = 0; t < T_; ++t) {
    floatx8 acc[8];  // acc[gi*2 + mt], bias folded in later
#pragma unroll
    for (int i = 0; i < 8; ++i)
#pragma unroll
      for (int r = 0; r < 8; ++r) acc[i][r] = 0.0f;

    if (khalf == 0) {
      kloop<0>(xA, xB, hA, hB, bw, hoff, acc);
    } else {
      kloop<1>(xA, xB, hA, hB, bw, hoff, acc);
      // publish partial sums to partner wave
#pragma unroll
      for (int i = 0; i < 8; ++i)
        *(floatx8*)&xch[p][i][lane * 8] = acc[i];
    }
    __syncthreads();

    if (khalf == 0) {
#pragma unroll
      for (int i = 0; i < 8; ++i) {
        floatx8 part = *(const floatx8*)&xch[p][i][lane * 8];
#pragma unroll
        for (int r = 0; r < 8; ++r) acc[i][r] += part[r];
      }
      // gate nonlinearities + state update; C/D layout: row = 8*half + r, col = n
#pragma unroll
      for (int mt = 0; mt < 2; ++mt) {
#pragma unroll
        for (int r = 0; r < 8; ++r) {
          float fv = sigmoid_(acc[0 + mt][r] + bias0);
          float iv = sigmoid_(acc[2 + mt][r] + bias1);
          float gv = tanhf(acc[4 + mt][r] + bias2);
          float ov = sigmoid_(acc[6 + mt][r] + bias3);
          float cv = gv * iv + cst[mt][r] * fv;
          cst[mt][r] = cv;
          float hv = tanhf(cv) * ov;
          const size_t idx = (size_t)(m0 + (mt << 4) + (half << 3) + r) * H_ + j0 + n;
          hb[idx] = (__bf16)hv;
          if (t == T_ - 1) hf[idx] = hv;
        }
      }
      if (t < T_ - 1) {  // speculative prefetch of next timestep's x rows
        __builtin_prefetch(xA + (size_t)B_ * D_, 0, 0);
        __builtin_prefetch(xB + (size_t)B_ * D_, 0, 0);
      }
    }
    xA += (size_t)B_ * D_;
    xB += (size_t)B_ * D_;

    // ---- grid-wide barrier (release h stores, acquire others') ----
    __threadfence();
    __syncthreads();
    if (threadIdx.x == 0) {
      __hip_atomic_fetch_add(cnt, 1u, __ATOMIC_RELEASE, __HIP_MEMORY_SCOPE_AGENT);
      const unsigned target = (unsigned)(t + 1) * (unsigned)NWG;
      while (__hip_atomic_load(cnt, __ATOMIC_ACQUIRE, __HIP_MEMORY_SCOPE_AGENT) < target) {
        __builtin_amdgcn_s_sleep(2);
      }
    }
    __syncthreads();
  }
}

// ---------------- final projection: out = h_T @ Wph + bp ----------------
__global__ __launch_bounds__(256) void k_proj(const float* __restrict__ hfin,
                                              const float* __restrict__ Wph,
                                              const float* __restrict__ bp,
                                              float* __restrict__ out) {
  const int b = blockIdx.x;
  const int tid = threadIdx.x;
  float s[10];
#pragma unroll
  for (int c = 0; c < 10; ++c) s[c] = 0.0f;
  for (int j = tid; j < H_; j += 256) {
    float hj = hfin[(size_t)b * H_ + j];
    const float* wr = Wph + (size_t)j * 10;
#pragma unroll
    for (int c = 0; c < 10; ++c) s[c] += hj * wr[c];
  }
  __shared__ float red[256];
  for (int c = 0; c < 10; ++c) {
    red[tid] = s[c];
    __syncthreads();
    for (int off = 128; off > 0; off >>= 1) {
      if (tid < off) red[tid] += red[tid + off];
      __syncthreads();
    }
    if (tid == 0) out[b * 10 + c] = red[0] + bp[c];
    __syncthreads();
  }
}

// ---------------- host launcher ----------------
extern "C" void kernel_launch(void* const* d_in, const int* in_sizes, int n_in,
                              void* d_out, int out_size, void* d_ws, size_t ws_size,
                              hipStream_t stream) {
  (void)in_sizes; (void)n_in; (void)out_size; (void)ws_size;
  const float* x   = (const float*)d_in[0];
  const float* Wfx = (const float*)d_in[1];
  const float* Wix = (const float*)d_in[2];
  const float* Wgx = (const float*)d_in[3];
  const float* Wox = (const float*)d_in[4];
  const float* Wfh = (const float*)d_in[5];
  const float* Wih = (const float*)d_in[6];
  const float* Wgh = (const float*)d_in[7];
  const float* Woh = (const float*)d_in[8];
  const float* bfv = (const float*)d_in[9];
  const float* biv = (const float*)d_in[10];
  const float* bgv = (const float*)d_in[11];
  const float* bov = (const float*)d_in[12];
  const float* Wph = (const float*)d_in[13];
  const float* bp  = (const float*)d_in[14];

  char* ws = (char*)d_ws;
  unsigned* cnt = (unsigned*)(ws + OFF_CNT);
  __bf16*   hb  = (__bf16*)(ws + OFF_HB);
  float*    hfn = (float*)(ws + OFF_HF);
  __bf16*   xbb = (__bf16*)(ws + OFF_XB);
  __bf16*   Wt  = (__bf16*)(ws + OFF_WT);

  k_zero<<<(B_ * H_ + 255) / 256, 256, 0, stream>>>(cnt, hb);
  {
    size_t N = (size_t)T_ * B_ * D_;
    k_convx<<<(unsigned)((N + 255) / 256), 256, 0, stream>>>(x, xbb);
  }
  {
    size_t N = (size_t)4 * H_ * KTOT;
    k_buildWt<<<(unsigned)((N + 255) / 256), 256, 0, stream>>>(
        Wfx, Wix, Wgx, Wox, Wfh, Wih, Wgh, Woh, Wt);
  }
  k_lstm<<<NWG, WGT, 0, stream>>>(xbb, Wt, bfv, biv, bgv, bov, hb, hfn, cnt);
  k_proj<<<B_, 256, 0, stream>>>(hfn, Wph, bp, (float*)d_out);
}
